// SiglipAttention_8469675508014
// MI455X (gfx1250) — matrix operand order repaired and checked by reading, not compiled, not run on hardware
//
#include <hip/hip_runtime.h>
#include <hip/hip_bf16.h>

typedef __attribute__((ext_vector_type(16))) __bf16 v16bf;
typedef __attribute__((ext_vector_type(8)))  __bf16 v8bf;
typedef __attribute__((ext_vector_type(4)))  __bf16 v4bf;
typedef __attribute__((ext_vector_type(2)))  __bf16 v2bf;
typedef __attribute__((ext_vector_type(8)))  float  v8f;
typedef __attribute__((ext_vector_type(4)))  float  v4f;
typedef __attribute__((ext_vector_type(4)))  unsigned int u32x4;
typedef __attribute__((ext_vector_type(8)))  int i32x8;
typedef __attribute__((ext_vector_type(4)))  int i32x4;

static constexpr int B_  = 32;
static constexpr int S_  = 729;
static constexpr int D_  = 1152;
static constexpr int H_  = 16;
static constexpr int HD  = 72;
static constexpr int HDP = 96;            // hd padded to 3 x WMMA-K(32)
static constexpr int M1  = B_ * S_;       // 23328
static constexpr int N1  = 3 * D_;        // 3456
static constexpr int K1  = D_;            // 1152

#define WMMA_BF16(a, b, c) \
  __builtin_amdgcn_wmma_f32_16x16x32_bf16(false, (a), false, (b), (short)0, (c), false, false)

#if defined(__has_builtin)
#  if __has_builtin(__builtin_amdgcn_tensor_load_to_lds)
#    define HAVE_TDM 1
#  endif
#endif
#ifndef HAVE_TDM
#  define HAVE_TDM 0
#endif

// generic pointer to an LDS object: low 32 bits are the LDS byte offset
__device__ inline unsigned lds_off(const void* p) {
  return (unsigned)(unsigned long long)p;
}

#if HAVE_TDM
// 2D bf16 tile DMA: global[row][col] (row stride = stride_elems, elements) ->
// LDS rows of tile_d0 elements (+ optional LDS padding). OOB rows/cols read 0.
// D# layout per CDNA5 ISA 8.3-8.6. Issue from ONE wave; wait TENSORcnt.
__device__ inline void tdm_load_2d(unsigned lds_byte_addr, const void* gptr,
                                   unsigned tile_d0, unsigned tile_d1,
                                   unsigned tensor_d0, unsigned tensor_d1,
                                   unsigned stride_elems, unsigned pad_en,
                                   unsigned pad_interval, unsigned pad_amount) {
  unsigned long long ga = (unsigned long long)gptr;
  u32x4 g0;
  g0[0] = 1u;                                            // count=1, user mode
  g0[1] = lds_byte_addr;                                 // lds_addr
  g0[2] = (unsigned)(ga & 0xffffffffu);                  // global_addr[31:0]
  g0[3] = (unsigned)((ga >> 32) & 0x01ffffffu) | (2u << 30);  // [56:32] | type=2
  i32x8 g1;
  g1[0] = (int)((1u << 16) | (pad_en << 20) | (pad_interval << 22) |
                (pad_amount << 25));                     // mask=0, data_size=2B
  g1[1] = (int)((tensor_d0 & 0xffffu) << 16);            // abar=0 | td0[15:0]
  g1[2] = (int)((tensor_d0 >> 16) | ((tensor_d1 & 0xffffu) << 16));
  g1[3] = (int)((tensor_d1 >> 16) | (tile_d0 << 16));
  g1[4] = (int)(tile_d1 & 0xffffu);                      // tile_d1 | tile_d2=0
  g1[5] = (int)stride_elems;                             // dim0_stride[31:0]
  g1[6] = 0;                                             // dim0_stride[47:32]
  g1[7] = 0;
  i32x4 z4; z4[0] = z4[1] = z4[2] = z4[3] = 0;
#if __clang_major__ >= 23
  i32x8 z8;
#pragma unroll
  for (int i = 0; i < 8; ++i) z8[i] = 0;
  __builtin_amdgcn_tensor_load_to_lds(g0, g1, z4, z4, z8, 0);
#else
  __builtin_amdgcn_tensor_load_to_lds(g0, g1, z4, z4, 0);
#endif
}
#endif

// ---- fragment loaders (ISA 7.12.2 layouts, wave32) -------------------------
__device__ inline v16bf ld_fragA(const __bf16* __restrict__ base, int ldk,
                                 int row0, int k0, int lane) {
  const __bf16* p = base + (row0 + (lane & 15)) * ldk + k0 + ((lane >> 4) * 8);
  v8bf a = *(const v8bf*)p;
  v8bf b = *(const v8bf*)(p + 16);
  v16bf r;
#pragma unroll
  for (int i = 0; i < 8; ++i) { r[i] = a[i]; r[i + 8] = b[i]; }
  return r;
}

__device__ inline v16bf ld_fragB(const __bf16* __restrict__ base, int ldk,
                                 int row0, int k0, int lane) {
  const __bf16* p = base + (row0 + (lane & 15)) * ldk + k0 + ((lane >> 4) * 8);
  v8bf a = *(const v8bf*)p;
  v8bf b = *(const v8bf*)(p + 16);
  v16bf r;
#pragma unroll
  for (int i = 0; i < 8; ++i) { r[i] = a[i]; r[i + 8] = b[i]; }
  return r;
}

// fp32 weight tile [32 x 128] -> bf16 LDS transposed [n][k]:
// each thread owns 4 consecutive k-rows x 4 columns -> 4x ds_store_b64
template <int NCOLS>
__device__ inline void load_wtile_T(const float* __restrict__ Bw, __bf16* sB,
                                    int LDK, int k0, int Nb, int tid,
                                    bool prefetch) {
  int kq = tid >> 5, c4 = tid & 31;
  int gn = Nb + c4 * 4;
  v4f vr[4];
#pragma unroll
  for (int i = 0; i < 4; ++i) {
    int gk = k0 + kq * 4 + i;
    vr[i] = *(const v4f*)(Bw + (size_t)gk * NCOLS + gn);
    if (prefetch)
      __builtin_prefetch(Bw + (size_t)(gk + 32) * NCOLS + gn, 0, 0);
  }
#pragma unroll
  for (int e = 0; e < 4; ++e) {
    v4bf o;
#pragma unroll
    for (int i = 0; i < 4; ++i) o[i] = (__bf16)vr[i][e];
    *(v4bf*)(sB + (c4 * 4 + e) * LDK + kq * 4) = o;
  }
}

// ---- kernel 1: fused QKV GEMM + bias + head-split to bf16 ------------------
__global__ __launch_bounds__(256) void qkv_gemm(
    const float* __restrict__ A, const float* __restrict__ Bw,
    const float* __restrict__ bias, unsigned short* __restrict__ qb_u,
    unsigned short* __restrict__ kb_u, unsigned short* __restrict__ vb_u) {
  constexpr int LDK = 40;
  __shared__ alignas(16) __bf16 sA[128 * LDK];
  __shared__ alignas(16) __bf16 sB[128 * LDK];   // stored transposed [n][k]
  __bf16* qb = (__bf16*)qb_u;
  __bf16* kb = (__bf16*)kb_u;
  __bf16* vb = (__bf16*)vb_u;

  const int tid = threadIdx.x, lane = tid & 31, w = tid >> 5;
  const int m0 = (w >> 2) * 64, n0 = (w & 3) * 32;
  const int Mb = blockIdx.y * 128, Nb = blockIdx.x * 128;

  v8f acc[4][2];
#pragma unroll
  for (int mt = 0; mt < 4; ++mt)
#pragma unroll
    for (int nt = 0; nt < 2; ++nt)
#pragma unroll
      for (int r = 0; r < 8; ++r) acc[mt][nt][r] = 0.f;

  for (int kb_ = 0; kb_ < K1 / 32; ++kb_) {
    const int k0 = kb_ * 32;
    __syncthreads();
    // A tile 128x32 fp32 -> bf16 LDS (clamped rows feed discarded acc rows)
#pragma unroll
    for (int j = 0; j < 4; ++j) {
      int f = tid + 256 * j;
      int r = f >> 3, c4 = f & 7;
      int gm = Mb + r;
      gm = gm < M1 ? gm : M1 - 1;
      v4f v = *(const v4f*)(A + (size_t)gm * K1 + k0 + c4 * 4);
      v4bf o;
#pragma unroll
      for (int e = 0; e < 4; ++e) o[e] = (__bf16)v[e];
      *(v4bf*)(sA + r * LDK + c4 * 4) = o;
    }
    // B tile 32x128 fp32 -> bf16 transposed, packed b64 stores
    load_wtile_T<N1>(Bw, sB, LDK, k0, Nb, tid, kb_ + 1 < K1 / 32);
    __syncthreads();
    v16bf af[4], bfr[2];
#pragma unroll
    for (int mt = 0; mt < 4; ++mt) af[mt] = ld_fragA(sA, LDK, m0 + mt * 16, 0, lane);
#pragma unroll
    for (int nt = 0; nt < 2; ++nt) bfr[nt] = ld_fragB(sB, LDK, n0 + nt * 16, 0, lane);
#pragma unroll
    for (int mt = 0; mt < 4; ++mt)
#pragma unroll
      for (int nt = 0; nt < 2; ++nt)
        acc[mt][nt] = WMMA_BF16(af[mt], bfr[nt], acc[mt][nt]);
  }

  const int hi = lane >> 4, nl = lane & 15;
#pragma unroll
  for (int mt = 0; mt < 4; ++mt)
#pragma unroll
    for (int nt = 0; nt < 2; ++nt)
#pragma unroll
      for (int r = 0; r < 8; ++r) {
        int m = Mb + m0 + mt * 16 + r + hi * 8;
        if (m >= M1) continue;
        int n = Nb + n0 + nt * 16 + nl;
        float val = acc[mt][nt][r] + bias[n];
        int part = n / D_;
        int rem = n - part * D_;
        int h = rem / HD, d = rem - h * HD;
        int bb = m / S_, s = m - bb * S_;
        __bf16* dst = (part == 0) ? qb : (part == 1) ? kb : vb;
        dst[(((size_t)bb * H_ + h) * S_ + s) * HDP + d] = (__bf16)val;
      }
}

// ---- kernel 2: flash attention per (b, h, q-tile of 128) -------------------
__global__ __launch_bounds__(256) void attn_kernel(
    const unsigned short* __restrict__ qb_u, const unsigned short* __restrict__ kb_u,
    const unsigned short* __restrict__ vb_u, unsigned short* __restrict__ out_u) {
  constexpr int QSTR = 96;   // TDM tiles: dense rows (192B, 16B-aligned)
  constexpr int PSTR = 136;  // 128 + 8
  constexpr int VSTR = 136;
  constexpr int SSTR = 132;  // fp32 scores
  constexpr int NKT = 6;     // ceil(S_/128)
  __shared__ alignas(16) __bf16 sQ[128 * QSTR];
  __shared__ alignas(16) __bf16 sK0[128 * QSTR];   // double-buffered K tiles
  __shared__ alignas(16) __bf16 sK1[128 * QSTR];
  __shared__ alignas(16) __bf16 sVt[96 * VSTR];    // V transposed: [d][k-pos]
  __shared__ alignas(16) __bf16 sP[128 * PSTR];
  __shared__ alignas(16) float sS[128 * SSTR];
  __shared__ float m_run[128], l_run[128], rowFactor[128];

  const __bf16* qb = (const __bf16*)qb_u;
  const __bf16* kb = (const __bf16*)kb_u;
  const __bf16* vb = (const __bf16*)vb_u;
  __bf16* attnout = (__bf16*)out_u;

  const int tid = threadIdx.x, lane = tid & 31, w = tid >> 5;
  const int hi = lane >> 4, nl = lane & 15;
  const int qt0 = blockIdx.x * 128;
  const int h = blockIdx.y, b = blockIdx.z;
  const size_t headBase = ((size_t)b * H_ + h) * S_ * HDP;

  // prologue: DMA Q tile and K tile 0 (OOB rows beyond S_ read as zero)
#if HAVE_TDM
  if (tid < 32) {
    tdm_load_2d(lds_off(sQ), qb + headBase + (size_t)qt0 * HDP,
                HDP, 128, HDP, (unsigned)(S_ - qt0), HDP, 0, 0, 0);
    tdm_load_2d(lds_off(sK0), kb + headBase,
                HDP, 128, HDP, (unsigned)S_, HDP, 0, 0, 0);
  }
#else
#pragma unroll
  for (int j = 0; j < 6; ++j) {
    int f = tid + 256 * j;
    int r = f / 12, c8 = f % 12;
    int s = qt0 + r;
    v8bf v;
#pragma unroll
    for (int e = 0; e < 8; ++e) v[e] = (__bf16)0.f;
    if (s < S_) v = *(const v8bf*)(qb + headBase + (size_t)s * HDP + c8 * 8);
    *(v8bf*)(sQ + r * QSTR + c8 * 8) = v;
  }
#endif

  if (tid < 128) { m_run[tid] = -1e30f; l_run[tid] = 0.f; }

  v8f oacc[6];
#pragma unroll
  for (int nt = 0; nt < 6; ++nt)
#pragma unroll
    for (int r = 0; r < 8; ++r) oacc[nt][r] = 0.f;

  const float scale = 0.11785113019775793f;  // 72^-0.5

  for (int kt = 0; kt < NKT; ++kt) {
    const int kt0 = kt * 128;
    __bf16* bufK = (kt & 1) ? sK1 : sK0;
    __bf16* nxtK = (kt & 1) ? sK0 : sK1;
    __syncthreads();
    // pipeline: DMA K tile kt+1 into the other buffer while we work on kt
#if HAVE_TDM
    if (tid < 32 && kt + 1 < NKT) {
      tdm_load_2d(lds_off(nxtK), kb + headBase + (size_t)(kt0 + 128) * HDP,
                  HDP, 128, HDP, (unsigned)(S_ - (kt0 + 128)), HDP, 0, 0, 0);
    }
#else
#pragma unroll
    for (int j = 0; j < 6; ++j) {
      int f = tid + 256 * j;
      int r = f / 12, c8 = f % 12;
      int s = kt0 + r;
      v8bf v;
#pragma unroll
      for (int e = 0; e < 8; ++e) v[e] = (__bf16)0.f;
      if (s < S_) v = *(const v8bf*)(kb + headBase + (size_t)s * HDP + c8 * 8);
      *(v8bf*)(bufK + r * QSTR + c8 * 8) = v;
    }
#endif
    // V tile -> sVt [d][k-pos]; 2 rows per thread -> packed b32 stores
    // (clamp is safe: P==0 for masked columns)
#pragma unroll
    for (int j = 0; j < 3; ++j) {
      int f = tid + 256 * j;          // 768 = 64 row-pairs x 12 col-groups
      int r2 = f / 12, c8 = f % 12;
      int s0 = kt0 + 2 * r2, s1 = s0 + 1;
      s0 = s0 < S_ ? s0 : S_ - 1;
      s1 = s1 < S_ ? s1 : S_ - 1;
      v8bf v0 = *(const v8bf*)(vb + headBase + (size_t)s0 * HDP + c8 * 8);
      v8bf v1 = *(const v8bf*)(vb + headBase + (size_t)s1 * HDP + c8 * 8);
#pragma unroll
      for (int e = 0; e < 8; ++e) {
        v2bf p; p[0] = v0[e]; p[1] = v1[e];
        *(v2bf*)(sVt + (c8 * 8 + e) * VSTR + 2 * r2) = p;
      }
    }
#if HAVE_TDM
    if (tid < 32) {
      if (kt + 1 < NKT) __builtin_amdgcn_s_wait_tensorcnt(1);  // kt done, kt+1 flying
      else              __builtin_amdgcn_s_wait_tensorcnt(0);
    }
#endif
    __syncthreads();

    // scores S = Q @ K^T : wave grid 2x4, each wave 64x32
    {
      const int m0 = (w >> 2) * 64, n0 = (w & 3) * 32;
      v8f acc[4][2];
#pragma unroll
      for (int mt = 0; mt < 4; ++mt)
#pragma unroll
        for (int nt = 0; nt < 2; ++nt)
#pragma unroll
          for (int r = 0; r < 8; ++r) acc[mt][nt][r] = 0.f;
#pragma unroll
      for (int ks = 0; ks < 3; ++ks) {
        v16bf af[4], bfr[2];
#pragma unroll
        for (int mt = 0; mt < 4; ++mt)
          af[mt] = ld_fragA(sQ, QSTR, m0 + mt * 16, ks * 32, lane);
#pragma unroll
        for (int nt = 0; nt < 2; ++nt)
          bfr[nt] = ld_fragB(bufK, QSTR, n0 + nt * 16, ks * 32, lane);
#pragma unroll
        for (int mt = 0; mt < 4; ++mt)
#pragma unroll
          for (int nt = 0; nt < 2; ++nt)
            acc[mt][nt] = WMMA_BF16(af[mt], bfr[nt], acc[mt][nt]);
      }
#pragma unroll
      for (int mt = 0; mt < 4; ++mt)
#pragma unroll
        for (int nt = 0; nt < 2; ++nt)
#pragma unroll
          for (int r = 0; r < 8; ++r) {
            int m = m0 + mt * 16 + r + hi * 8;
            int n = n0 + nt * 16 + nl;
            float val = (kt0 + n < S_) ? acc[mt][nt][r] * scale : -1e30f;
            sS[m * SSTR + n] = val;
          }
    }
    __syncthreads();

    // online softmax: 2 threads per row, combine halves via shfl
    {
      int row = tid >> 1, half = tid & 1;
      const float* srow = sS + row * SSTR + half * 64;
      float mx = -1e30f;
      for (int j2 = 0; j2 < 64; ++j2) mx = fmaxf(mx, srow[j2]);
      mx = fmaxf(mx, __shfl_xor(mx, 1));
      float m_old = m_run[row];
      float m_new = fmaxf(m_old, mx);
      float sum = 0.f;
      __bf16* prow = sP + row * PSTR + half * 64;
      for (int j2 = 0; j2 < 64; ++j2) {
        float p = __expf(srow[j2] - m_new);
        sum += p;
        prow[j2] = (__bf16)p;
      }
      sum += __shfl_xor(sum, 1);
      if (half == 0) {
        float fct = __expf(m_old - m_new);
        rowFactor[row] = fct;
        l_run[row] = l_run[row] * fct + sum;
        m_run[row] = m_new;
      }
    }
    __syncthreads();

    // O rescale + O += P @ V : wave w owns q-rows [w*16, w*16+16)
    {
#pragma unroll
      for (int r = 0; r < 8; ++r) {
        float fct = rowFactor[w * 16 + r + hi * 8];
#pragma unroll
        for (int nt = 0; nt < 6; ++nt) oacc[nt][r] *= fct;
      }
#pragma unroll
      for (int ks = 0; ks < 4; ++ks) {
        v16bf pA = ld_fragA(sP, PSTR, w * 16, ks * 32, lane);
#pragma unroll
        for (int nt = 0; nt < 6; ++nt) {
          v16bf vB = ld_fragB(sVt, VSTR, nt * 16, ks * 32, lane);
          oacc[nt] = WMMA_BF16(pA, vB, oacc[nt]);
        }
      }
    }
  }
  __syncthreads();

  // write O / l  -> attn output [B, S, H*HD] bf16
#pragma unroll
  for (int nt = 0; nt < 6; ++nt)
#pragma unroll
    for (int r = 0; r < 8; ++r) {
      int row = w * 16 + r + hi * 8;
      int s = qt0 + row;
      int d = nt * 16 + nl;
      if (s < S_ && d < HD) {
        float invl = 1.0f / l_run[row];
        attnout[((size_t)b * S_ + s) * D_ + h * HD + d] = (__bf16)(oacc[nt][r] * invl);
      }
    }
}

// ---- kernel 3: output projection GEMM --------------------------------------
__global__ __launch_bounds__(256) void out_gemm(
    const unsigned short* __restrict__ A_u, const float* __restrict__ Bw,
    const float* __restrict__ bias, float* __restrict__ Out) {
  constexpr int LDK = 40;
  constexpr int NK = D_ / 32;
  __shared__ alignas(16) __bf16 sA0[128 * LDK];   // double-buffered TDM A tiles
  __shared__ alignas(16) __bf16 sA1[128 * LDK];
  __shared__ alignas(16) __bf16 sB[128 * LDK];
  const __bf16* A = (const __bf16*)A_u;

  const int tid = threadIdx.x, lane = tid & 31, w = tid >> 5;
  const int m0 = (w >> 2) * 64, n0 = (w & 3) * 32;
  const int Mb = blockIdx.y * 128, Nb = blockIdx.x * 128;

  v8f acc[4][2];
#pragma unroll
  for (int mt = 0; mt < 4; ++mt)
#pragma unroll
    for (int nt = 0; nt < 2; ++nt)
#pragma unroll
      for (int r = 0; r < 8; ++r) acc[mt][nt][r] = 0.f;

  // A tiles via TDM with LDS padding: every 16 DWORDs (32 bf16) add 4 DWORDs
  // (8 bf16) -> LDS row stride 40. OOB rows (m >= M1) read zero.
#if HAVE_TDM
  if (tid < 32) {
    tdm_load_2d(lds_off(sA0), A + (size_t)Mb * D_,
                32, 128, 32, (unsigned)(M1 - Mb), D_, 1, 3, 3);
  }
#endif

  for (int kb_ = 0; kb_ < NK; ++kb_) {
    const int k0 = kb_ * 32;
    __bf16* bufA = (kb_ & 1) ? sA1 : sA0;
    __bf16* nxtA = (kb_ & 1) ? sA0 : sA1;
    __syncthreads();
#if HAVE_TDM
    if (tid < 32 && kb_ + 1 < NK) {
      tdm_load_2d(lds_off(nxtA), A + (size_t)Mb * D_ + (k0 + 32),
                  32, 128, 32, (unsigned)(M1 - Mb), D_, 1, 3, 3);
    }
#else
#pragma unroll
    for (int j = 0; j < 2; ++j) {
      int f = tid + 256 * j;
      int r = f >> 2, c8 = f & 3;
      int gm = Mb + r;
      gm = gm < M1 ? gm : M1 - 1;
      v8bf v = *(const v8bf*)(A + (size_t)gm * D_ + k0 + c8 * 8);
      *(v8bf*)(bufA + r * LDK + c8 * 8) = v;
    }
#endif
    // B tile 32x128 fp32 -> bf16 transposed, packed b64 stores
    load_wtile_T<D_>(Bw, sB, LDK, k0, Nb, tid, kb_ + 1 < NK);
#if HAVE_TDM
    if (tid < 32) {
      if (kb_ + 1 < NK) __builtin_amdgcn_s_wait_tensorcnt(1);
      else              __builtin_amdgcn_s_wait_tensorcnt(0);
    }
#endif
    __syncthreads();
    v16bf af[4], bfr[2];
#pragma unroll
    for (int mt = 0; mt < 4; ++mt) af[mt] = ld_fragA(bufA, LDK, m0 + mt * 16, 0, lane);
#pragma unroll
    for (int nt = 0; nt < 2; ++nt) bfr[nt] = ld_fragB(sB, LDK, n0 + nt * 16, 0, lane);
#pragma unroll
    for (int mt = 0; mt < 4; ++mt)
#pragma unroll
      for (int nt = 0; nt < 2; ++nt)
        acc[mt][nt] = WMMA_BF16(af[mt], bfr[nt], acc[mt][nt]);
  }

  const int hi = lane >> 4, nl = lane & 15;
#pragma unroll
  for (int mt = 0; mt < 4; ++mt)
#pragma unroll
    for (int nt = 0; nt < 2; ++nt)
#pragma unroll
      for (int r = 0; r < 8; ++r) {
        int m = Mb + m0 + mt * 16 + r + hi * 8;
        if (m >= M1) continue;
        int n = Nb + n0 + nt * 16 + nl;
        Out[(size_t)m * D_ + n] = acc[mt][nt][r] + bias[n];
      }
}

// ---- launch ---------------------------------------------------------------
extern "C" void kernel_launch(void* const* d_in, const int* in_sizes, int n_in,
                              void* d_out, int out_size, void* d_ws, size_t ws_size,
                              hipStream_t stream) {
  (void)in_sizes; (void)n_in; (void)out_size; (void)ws_size;
  const float* hidden = (const float*)d_in[0];
  const float* w_qkv  = (const float*)d_in[1];
  const float* b_qkv  = (const float*)d_in[2];
  const float* w_out  = (const float*)d_in[3];
  const float* b_out  = (const float*)d_in[4];
  float* out = (float*)d_out;

  const size_t headElems = (size_t)B_ * H_ * S_ * HDP;  // per q/k/v buffer
  unsigned short* qb = (unsigned short*)d_ws;
  unsigned short* kb = qb + headElems;
  unsigned short* vb = kb + headElems;
  unsigned short* attn = vb + headElems;                // [M1, D_] bf16

  // zero q/k/v so the hd 72->96 pad columns are 0 (graph-capture-safe)
  hipMemsetAsync(qb, 0, headElems * 3 * sizeof(unsigned short), stream);

  dim3 blk(256);
  qkv_gemm<<<dim3(N1 / 128, (M1 + 127) / 128), blk, 0, stream>>>(
      hidden, w_qkv, b_qkv, qb, kb, vb);
  attn_kernel<<<dim3((S_ + 127) / 128, H_, B_), blk, 0, stream>>>(qb, kb, vb, attn);
  out_gemm<<<dim3(D_ / 128, (M1 + 127) / 128), blk, 0, stream>>>(
      attn, w_out, b_out, out);
}
